// I2Block_12154757448425
// MI455X (gfx1250) — matrix operand order repaired and checked by reading, not compiled, not run on hardware
//
#include <hip/hip_runtime.h>
#include <hip/hip_bf16.h>

// ---------------- Problem constants ----------------
#define D_MODEL   256
#define D_STATE   16
#define D_CONV    4
#define D_INNER   512
#define DT_RANK   16
#define NROWS     8192              // B*L = 8 * 1024
#define SEQ_L     1024
#define NBATCH    8
#define NCHUNK    16                // scan chunks per sequence
#define TCHUNK    (SEQ_L / NCHUNK)  // 64 timesteps per chunk
#define XDBL_W    (DT_RANK + 2 * D_STATE)   // 48

typedef __attribute__((ext_vector_type(16))) __bf16 v16bf;
typedef __attribute__((ext_vector_type(8)))  float  v8f;

// ---------------- WMMA tile loaders (CDNA5 wave32 layouts) ----------------
// A-matrix 16x32 bf16: lanes 0-15 rows 0-15 {K 0-7 | K 16-23}, lanes 16-31 {K 8-15 | K 24-31}
__device__ __forceinline__ v16bf load_a_f32(const float* A, int lda, int row0,
                                            int k0, int Kreal, int lane) {
  int r  = row0 + (lane & 15);
  int kh = (lane >> 4) * 8;
  const float* p = A + (size_t)r * lda + k0 + kh;
  bool v0 = (k0 + kh)      < Kreal;
  bool v1 = (k0 + kh + 16) < Kreal;
  v16bf a;
#pragma unroll
  for (int i = 0; i < 8; ++i) {
    a[i]     = v0 ? (__bf16)p[i]      : (__bf16)0.0f;
    a[i + 8] = v1 ? (__bf16)p[i + 16] : (__bf16)0.0f;
  }
  return a;
}

__device__ __forceinline__ v16bf load_a_bf16(const __bf16* A, int lda, int row0,
                                             int k0, int Kreal, int lane) {
  int r  = row0 + (lane & 15);
  int kh = (lane >> 4) * 8;
  const __bf16* p = A + (size_t)r * lda + k0 + kh;
  bool v0 = (k0 + kh)      < Kreal;
  bool v1 = (k0 + kh + 16) < Kreal;
  v16bf a;
#pragma unroll
  for (int i = 0; i < 8; ++i) {
    a[i]     = v0 ? p[i]      : (__bf16)0.0f;
    a[i + 8] = v1 ? p[i + 16] : (__bf16)0.0f;
  }
  return a;
}

// B-matrix 32x16 bf16 from weight W[N][K] (row major). B[k][n] = W[n0+n][k0+k].
// lanes 0-15: cols, {K 0-7 | K 16-23}; lanes 16-31: {K 8-15 | K 24-31}.
__device__ __forceinline__ v16bf load_b_w(const __bf16* W, int ldw, int n0,
                                          int k0, int Kreal, int lane) {
  int n  = n0 + (lane & 15);
  int kh = (lane >> 4) * 8;
  const __bf16* p = W + (size_t)n * ldw + k0 + kh;
  bool v0 = (k0 + kh) < Kreal;
  bool v1 = (k0 + kh + 16) < Kreal;
  v16bf b;
#pragma unroll
  for (int i = 0; i < 16; ++i) b[i] = (i < 8 ? v0 : v1) ? p[i + (i & 8)] : (__bf16)0.0f;
  return b;
}

__device__ __forceinline__ float dev_sigmoid(float x) { return 1.0f / (1.0f + __expf(-x)); }
__device__ __forceinline__ float dev_softplus(float x) {
  return (x > 20.0f) ? x : log1pf(__expf(x));
}

// ---------------- Kernels ----------------
__global__ void k_cvt_bf16(__bf16* __restrict__ dst, const float* __restrict__ src, int n) {
  int i = blockIdx.x * blockDim.x + threadIdx.x;
  if (i < n) dst[i] = (__bf16)src[i];
}

// xz[8192,1024] = X[8192,256](f32) * Win[1024,256]^T   (bf16 WMMA, f32 accum)
__global__ void k_in_proj(const float* __restrict__ X, const __bf16* __restrict__ W,
                          float* __restrict__ Y) {
  const int K = D_MODEL, N = 2 * D_INNER;
  int lane = threadIdx.x & 31, wave = threadIdx.x >> 5;
  int row0 = blockIdx.x * 64 + wave * 16;
  int n0   = blockIdx.y * 64;
  v8f acc[4];
#pragma unroll
  for (int j = 0; j < 4; ++j) acc[j] = 0.0f;
  for (int k0 = 0; k0 < K; k0 += 32) {
    v16bf a = load_a_f32(X, K, row0, k0, K, lane);
#pragma unroll
    for (int j = 0; j < 4; ++j) {
      v16bf b = load_b_w(W, K, n0 + 16 * j, k0, K, lane);
      acc[j] = __builtin_amdgcn_wmma_f32_16x16x32_bf16(false, a, false, b,
                                                       (short)0, acc[j], false, false);
    }
  }
  int col = lane & 15;
  int rb  = row0 + ((lane >> 4) ? 8 : 0);
#pragma unroll
  for (int j = 0; j < 4; ++j)
#pragma unroll
    for (int r = 0; r < 8; ++r)
      Y[(size_t)(rb + r) * N + n0 + 16 * j + col] = acc[j][r];
}

// Causal depthwise conv (k=4) + bias + SiLU. One thread per output element (row,d).
__global__ void k_conv(const float* __restrict__ xz, const float* __restrict__ cw,
                       const float* __restrict__ cb, float* __restrict__ xc,
                       __bf16* __restrict__ xcb) {
  size_t gid = (size_t)blockIdx.x * blockDim.x + threadIdx.x;   // 0 .. 8192*512-1
  int d   = (int)(gid & (D_INNER - 1));
  int row = (int)(gid >> 9);                 // b*1024 + t
  int t   = row & (SEQ_L - 1);
  const float* col = xz + (size_t)row * (2 * D_INNER) + d;      // x_in = xz[..., :512]
  const size_t rstride = 2 * D_INNER;
  float x3 = col[0];
  float x2 = (t >= 1) ? col[-(ptrdiff_t)rstride]     : 0.0f;
  float x1 = (t >= 2) ? col[-(ptrdiff_t)(2*rstride)] : 0.0f;
  float x0 = (t >= 3) ? col[-(ptrdiff_t)(3*rstride)] : 0.0f;
  float v = cw[d*4+0]*x0 + cw[d*4+1]*x1 + cw[d*4+2]*x2 + cw[d*4+3]*x3 + cb[d];
  v = v * dev_sigmoid(v);                    // SiLU
  xc [(size_t)row * D_INNER + d] = v;
  xcb[(size_t)row * D_INNER + d] = (__bf16)v;
}

// x_dbl[8192,48] = xc(bf16)[8192,512] * Wxp[48,512]^T
__global__ void k_x_proj(const __bf16* __restrict__ A, const __bf16* __restrict__ W,
                         float* __restrict__ Y) {
  const int K = D_INNER, N = XDBL_W;                   // 48
  int lane = threadIdx.x & 31, wave = threadIdx.x >> 5;
  int row0 = blockIdx.x * 64 + wave * 16;
  v8f acc[3];
#pragma unroll
  for (int j = 0; j < 3; ++j) acc[j] = 0.0f;
  for (int k0 = 0; k0 < K; k0 += 32) {
    v16bf a = load_a_bf16(A, K, row0, k0, K, lane);
#pragma unroll
    for (int j = 0; j < 3; ++j) {
      v16bf b = load_b_w(W, K, 16 * j, k0, K, lane);
      acc[j] = __builtin_amdgcn_wmma_f32_16x16x32_bf16(false, a, false, b,
                                                       (short)0, acc[j], false, false);
    }
  }
  int col = lane & 15;
  int rb  = row0 + ((lane >> 4) ? 8 : 0);
#pragma unroll
  for (int j = 0; j < 3; ++j)
#pragma unroll
    for (int r = 0; r < 8; ++r)
      Y[(size_t)(rb + r) * N + 16 * j + col] = acc[j][r];
}

// delta[8192,512] = softplus( dt[8192,16] * Wdt[512,16]^T + dt_b )
// dt = x_dbl[:, :16] (lda=48); K zero-padded 16 -> 32 in loaders.
__global__ void k_dt_proj(const float* __restrict__ XDBL, const __bf16* __restrict__ W,
                          const float* __restrict__ dtb, float* __restrict__ DL) {
  int lane = threadIdx.x & 31, wave = threadIdx.x >> 5;
  int row0 = blockIdx.x * 64 + wave * 16;
  int n0   = blockIdx.y * 64;
  v8f acc[4];
#pragma unroll
  for (int j = 0; j < 4; ++j) acc[j] = 0.0f;
  {
    v16bf a = load_a_f32(XDBL, XDBL_W, row0, 0, DT_RANK, lane);
#pragma unroll
    for (int j = 0; j < 4; ++j) {
      v16bf b = load_b_w(W, DT_RANK, n0 + 16 * j, 0, DT_RANK, lane);
      acc[j] = __builtin_amdgcn_wmma_f32_16x16x32_bf16(false, a, false, b,
                                                       (short)0, acc[j], false, false);
    }
  }
  int col = lane & 15;
  int rb  = row0 + ((lane >> 4) ? 8 : 0);
#pragma unroll
  for (int j = 0; j < 4; ++j) {
    int n = n0 + 16 * j + col;
    float bv = dtb[n];
#pragma unroll
    for (int r = 0; r < 8; ++r)
      DL[(size_t)(rb + r) * D_INNER + n] = dev_softplus(acc[j][r] + bv);
  }
}

// Cooperative stage of one chunk's B/C rows (TCHUNK x 32 floats) into LDS.
__device__ __forceinline__ void stage_bc(float (*sBC)[2 * D_STATE],
                                         const float* __restrict__ XDBL,
                                         int b, int t0) {
  for (int i = threadIdx.x; i < TCHUNK * 2 * D_STATE; i += D_INNER) {
    int tt = i >> 5, j = i & 31;
    sBC[tt][j] = XDBL[(size_t)(b * SEQ_L + t0 + tt) * XDBL_W + DT_RANK + j];
  }
  __syncthreads();
}

// ---- Chunked parallel scan: h_t = dA_t * h_{t-1} + dBu_t (associative) ----
// Pass 1: per (b,chunk,d) compute chunk transition: aProd[n] = prod dA,
//         s[n] = chunk-local scan from h=0. One lane per channel.
__global__ void k_scan_part(const float* __restrict__ DL, const float* __restrict__ xc,
                            const float* __restrict__ XDBL, const float* __restrict__ A_log,
                            float* __restrict__ aprod, float* __restrict__ ssum) {
  __shared__ float sBC[TCHUNK][2 * D_STATE];           // 8 KB
  int b = blockIdx.x >> 4;          // /NCHUNK
  int c = blockIdx.x & (NCHUNK - 1);
  int d = threadIdx.x;              // 0..511
  int t0 = c * TCHUNK;
  stage_bc(sBC, XDBL, b, t0);
  float An[D_STATE], aP[D_STATE], h[D_STATE];
#pragma unroll
  for (int n = 0; n < D_STATE; ++n) {
    An[n] = -__expf(A_log[d * D_STATE + n]);
    aP[n] = 1.0f;
    h[n]  = 0.0f;
  }
  for (int tt = 0; tt < TCHUNK; ++tt) {
    size_t row = (size_t)(b * SEQ_L + t0 + tt);
    float dl = DL[row * D_INNER + d];
    float x  = xc[row * D_INNER + d];
    float dx = dl * x;
#pragma unroll
    for (int n = 0; n < D_STATE; ++n) {
      float dA = __expf(dl * An[n]);
      aP[n] *= dA;
      h[n] = dA * h[n] + dx * sBC[tt][n];
    }
  }
  size_t base = ((size_t)(b * NCHUNK + c) * D_INNER + d) * D_STATE;
#pragma unroll
  for (int n = 0; n < D_STATE; ++n) {
    aprod[base + n] = aP[n];
    ssum [base + n] = h[n];
  }
}

// Pass 2: per (b,d,n) combine chunk transitions sequentially (16 steps),
//         emit the state entering each chunk.
__global__ void k_scan_combine(const float* __restrict__ aprod,
                               const float* __restrict__ ssum,
                               float* __restrict__ hinit) {
  int gid = blockIdx.x * blockDim.x + threadIdx.x;   // 0 .. 8*512*16-1
  int b  = gid >> 13;                                // /(512*16)
  int dn = gid & 8191;                               // d*16 + n
  float h = 0.0f;
  for (int c = 0; c < NCHUNK; ++c) {
    size_t idx = ((size_t)(b * NCHUNK + c) << 13) + dn;
    hinit[idx] = h;
    h = aprod[idx] * h + ssum[idx];
  }
}

// Pass 3: re-run each chunk from its entry state; fused epilogue:
//         y = (scan_y + xc*D) * SiLU(z), stored bf16 for out_proj.
__global__ void k_scan_final(const float* __restrict__ DL, const float* __restrict__ xc,
                             const float* __restrict__ XDBL, const float* __restrict__ xz,
                             const float* __restrict__ A_log, const float* __restrict__ Dsk,
                             const float* __restrict__ hinit, __bf16* __restrict__ Yb) {
  __shared__ float sBC[TCHUNK][2 * D_STATE];           // 8 KB
  int b = blockIdx.x >> 4;
  int c = blockIdx.x & (NCHUNK - 1);
  int d = threadIdx.x;
  int t0 = c * TCHUNK;
  stage_bc(sBC, XDBL, b, t0);
  float An[D_STATE], h[D_STATE];
  size_t base = ((size_t)(b * NCHUNK + c) * D_INNER + d) * D_STATE;
#pragma unroll
  for (int n = 0; n < D_STATE; ++n) {
    An[n] = -__expf(A_log[d * D_STATE + n]);
    h[n]  = hinit[base + n];
  }
  float Dd = Dsk[d];
  for (int tt = 0; tt < TCHUNK; ++tt) {
    size_t row = (size_t)(b * SEQ_L + t0 + tt);
    float dl = DL[row * D_INNER + d];
    float x  = xc[row * D_INNER + d];
    float dx = dl * x;
    float y  = 0.0f;
#pragma unroll
    for (int n = 0; n < D_STATE; ++n) {
      float dA = __expf(dl * An[n]);
      h[n] = dA * h[n] + dx * sBC[tt][n];
      y += h[n] * sBC[tt][D_STATE + n];
    }
    y += x * Dd;
    float z = xz[row * (2 * D_INNER) + D_INNER + d];
    Yb[row * D_INNER + d] = (__bf16)(y * (z * dev_sigmoid(z)));
  }
}

// out[8192,256] = y(bf16)[8192,512] * Wout[256,512]^T + residual x
__global__ void k_out_proj(const __bf16* __restrict__ A, const __bf16* __restrict__ W,
                           const float* __restrict__ RES, float* __restrict__ OUT) {
  const int K = D_INNER, N = D_MODEL;
  int lane = threadIdx.x & 31, wave = threadIdx.x >> 5;
  int row0 = blockIdx.x * 64 + wave * 16;
  int n0   = blockIdx.y * 64;
  v8f acc[4];
#pragma unroll
  for (int j = 0; j < 4; ++j) acc[j] = 0.0f;
  for (int k0 = 0; k0 < K; k0 += 32) {
    v16bf a = load_a_bf16(A, K, row0, k0, K, lane);
#pragma unroll
    for (int j = 0; j < 4; ++j) {
      v16bf b = load_b_w(W, K, n0 + 16 * j, k0, K, lane);
      acc[j] = __builtin_amdgcn_wmma_f32_16x16x32_bf16(false, a, false, b,
                                                       (short)0, acc[j], false, false);
    }
  }
  int col = lane & 15;
  int rb  = row0 + ((lane >> 4) ? 8 : 0);
#pragma unroll
  for (int j = 0; j < 4; ++j)
#pragma unroll
    for (int r = 0; r < 8; ++r) {
      size_t idx = (size_t)(rb + r) * N + n0 + 16 * j + col;
      OUT[idx] = acc[j][r] + RES[idx];
    }
}

// ---------------- Host launch ----------------
extern "C" void kernel_launch(void* const* d_in, const int* in_sizes, int n_in,
                              void* d_out, int out_size, void* d_ws, size_t ws_size,
                              hipStream_t stream) {
  const float* x         = (const float*)d_in[0];   // (2,4,1024,256) -> 8192x256
  const float* in_proj_w = (const float*)d_in[1];   // 1024x256
  const float* conv_w    = (const float*)d_in[2];   // 512x1x4
  const float* conv_b    = (const float*)d_in[3];   // 512
  const float* x_proj_w  = (const float*)d_in[4];   // 48x512
  const float* dt_proj_w = (const float*)d_in[5];   // 512x16
  const float* dt_proj_b = (const float*)d_in[6];   // 512
  const float* A_log     = (const float*)d_in[7];   // 512x16
  const float* D_skip    = (const float*)d_in[8];   // 512
  const float* out_proj_w= (const float*)d_in[9];   // 256x512
  float* out = (float*)d_out;

  char* ws = (char*)d_ws;
  size_t off = 0;
  __bf16* w_in  = (__bf16*)(ws + off); off += (size_t)1024 * 256 * 2;   // 512 KB
  __bf16* w_xp  = (__bf16*)(ws + off); off += (size_t)48 * 512 * 2;     // 48 KB
  __bf16* w_dt  = (__bf16*)(ws + off); off += (size_t)512 * 16 * 2;     // 16 KB
  __bf16* w_out = (__bf16*)(ws + off); off += (size_t)256 * 512 * 2;    // 256 KB
  float*  xz    = (float*) (ws + off); off += (size_t)NROWS * 1024 * 4; // 32 MB
  float*  xc    = (float*) (ws + off); off += (size_t)NROWS * 512 * 4;  // 16 MB
  __bf16* xcb   = (__bf16*)(ws + off); off += (size_t)NROWS * 512 * 2;  // 8 MB
  float*  xdbl  = (float*) (ws + off); off += (size_t)NROWS * 48 * 4;   // 1.5 MB
  float*  delta = (float*) (ws + off); off += (size_t)NROWS * 512 * 4;  // 16 MB
  __bf16* yb    = (__bf16*)(ws + off); off += (size_t)NROWS * 512 * 2;  // 8 MB
  const size_t scan_elems = (size_t)NBATCH * NCHUNK * D_INNER * D_STATE; // 1M
  float*  aprod = (float*) (ws + off); off += scan_elems * 4;           // 4 MB
  float*  ssum  = (float*) (ws + off); off += scan_elems * 4;           // 4 MB
  float*  hinit = (float*) (ws + off); off += scan_elems * 4;           // 4 MB
  (void)ws_size; (void)in_sizes; (void)n_in; (void)out_size;

  // weight conversion f32 -> bf16
  k_cvt_bf16<<<(1024 * 256 + 255) / 256, 256, 0, stream>>>(w_in,  in_proj_w,  1024 * 256);
  k_cvt_bf16<<<(48 * 512  + 255) / 256, 256, 0, stream>>>(w_xp,  x_proj_w,   48 * 512);
  k_cvt_bf16<<<(512 * 16  + 255) / 256, 256, 0, stream>>>(w_dt,  dt_proj_w,  512 * 16);
  k_cvt_bf16<<<(256 * 512 + 255) / 256, 256, 0, stream>>>(w_out, out_proj_w, 256 * 512);

  // pipeline
  k_in_proj <<<dim3(NROWS / 64, 1024 / 64), 128, 0, stream>>>(x, w_in, xz);
  k_conv    <<<(NROWS * D_INNER) / 256, 256, 0, stream>>>(xz, conv_w, conv_b, xc, xcb);
  k_x_proj  <<<dim3(NROWS / 64, 1), 128, 0, stream>>>(xcb, w_xp, xdbl);
  k_dt_proj <<<dim3(NROWS / 64, 512 / 64), 128, 0, stream>>>(xdbl, w_dt, dt_proj_b, delta);
  k_scan_part   <<<NBATCH * NCHUNK, D_INNER, 0, stream>>>(delta, xc, xdbl, A_log, aprod, ssum);
  k_scan_combine<<<(NBATCH * D_INNER * D_STATE) / 256, 256, 0, stream>>>(aprod, ssum, hinit);
  k_scan_final  <<<NBATCH * NCHUNK, D_INNER, 0, stream>>>(delta, xc, xdbl, xz, A_log,
                                                          D_skip, hinit, yb);
  k_out_proj<<<dim3(NROWS / 64, 256 / 64), 128, 0, stream>>>(yb, w_out, x, out);
}
